// FvTEncoder_88622355186280
// MI455X (gfx1250) — matrix operand order repaired and checked by reading, not compiled, not run on hardware
//
#include <hip/hip_runtime.h>
#include <math.h>

// ---------------------------------------------------------------------------
// Fully-fused FvT encoder for MI455X (gfx1250, wave32, WMMA).
// One workgroup = 16 events; all activations live in LDS (~269 KB of the
// 320 KB WGP LDS). All channel-mixing layers run on v_wmma_f32_16x16x32_bf16.
// Weights are converted f32->bf16 into d_ws by a prepass kernel and then
// streamed from L2 (all weights ~2.3MB bf16, L2 is 192MB).
// ---------------------------------------------------------------------------

#define SLOPEV 0.2291666666666667f
#define EPSV   1e-5f
#define SMEM_BYTES 274944

typedef __bf16 bf16_t;
typedef bf16_t bf16x16 __attribute__((ext_vector_type(16)));
typedef float  f32x8   __attribute__((ext_vector_type(8)));

union FragU { uint4 q[2]; bf16x16 v; };

__device__ __forceinline__ float bf2f(unsigned short u) {
  union { unsigned int i; float f; } v; v.i = ((unsigned int)u) << 16; return v.f;
}
__device__ __forceinline__ unsigned short f2bf(float f) {
  union { float f; unsigned int i; } v; v.f = f;
  unsigned int u = v.i;
  unsigned int r = u + 0x7FFFu + ((u >> 16) & 1u);           // RNE
  if ((u & 0x7F800000u) == 0x7F800000u) r = u;               // inf/nan
  return (unsigned short)(r >> 16);
}
__device__ __forceinline__ float nonlu(float x) { return x >= 0.f ? x : SLOPEV * x; }
__device__ __forceinline__ float dphi_wrap(float a) {
  const float PI_ = 3.14159265358979323846f, TP = 6.28318530717958647692f;
  float r = fmodf(a + PI_, TP);
  if (r < 0.f) r += TP;
  return r - PI_;
}

// A-matrix 16x32 bf16 fragment: lane holds row M=lane&15,
// K = k + half*16 + {0..15} (contiguous, matching frag_b).
__device__ __forceinline__ bf16x16 frag_a(const unsigned short* p) {
  FragU f;
  f.q[0] = *(const uint4*)(p);
  f.q[1] = *(const uint4*)(p + 8);
  return f.v;
}
// B-matrix 32x16 bf16 fragment: lane holds col N=lane&15, K = k + half*16 + {0..15}
// (contiguous along the weight row).
__device__ __forceinline__ bf16x16 frag_b(const unsigned short* p) {
  FragU f;
  f.q[0] = *(const uint4*)(p);
  f.q[1] = *(const uint4*)(p + 8);
  return f.v;
}

// Epilogue: y = (acc + bias)*bnScale + bnShift  (+resid)  (nonlu)  -> dst (,dst2)
__device__ __forceinline__ void epi_store(
    f32x8 acc, int oc, int mofs,
    const float* bias, const float* bng, const float* bnb,
    const float* bnm, const float* bnv,
    const unsigned short* resid, unsigned short* dst, unsigned short* dst2,
    bool act) {
  float b = bias ? bias[oc] : 0.f;
  float scale = 1.f, shift = 0.f;
  if (bng) { float s = bng[oc] / sqrtf(bnv[oc] + EPSV); scale = s; shift = bnb[oc] - bnm[oc] * s; }
#pragma unroll
  for (int r = 0; r < 8; ++r) {
    int tok = mofs + r;                   // D layout: lanes0-15 M=r, lanes16-31 M=r+8
    float v = (acc[r] + b) * scale + shift;
    if (resid) v += bf2f(resid[tok * 256 + oc]);
    if (act) v = nonlu(v);
    unsigned short h = f2bf(v);
    dst[tok * 256 + oc] = h;
    if (dst2) dst2[tok * 256 + oc] = h;
  }
}

// 16 tokens x 256 out-channels GEMM; A panel in LDS [16][Kt], W bf16 [256][Kt].
// 8 waves, each owns 2 oc-tiles of 16.
__device__ __forceinline__ void gemm16(
    const unsigned short* __restrict__ AP, int Kt,
    const unsigned short* __restrict__ W,
    const float* bias, const float* bng, const float* bnb,
    const float* bnm, const float* bnv,
    const unsigned short* resid, unsigned short* dst, unsigned short* dst2,
    bool act) {
  const int lane = threadIdx.x & 31;
  const int wave = threadIdx.x >> 5;
  const int col  = lane & 15;
  const int half = lane >> 4;
  const int oc0  = wave * 32;

  f32x8 acc0 = {};
  f32x8 acc1 = {};
  const unsigned short* ar = AP + col * Kt + half * 16;
  const unsigned short* b0 = W + (size_t)(oc0 + col) * Kt + half * 16;
  const unsigned short* b1 = W + (size_t)(oc0 + 16 + col) * Kt + half * 16;
  for (int k = 0; k < Kt; k += 32) {
    bf16x16 a  = frag_a(ar + k);
    bf16x16 w0 = frag_b(b0 + k);
    bf16x16 w1 = frag_b(b1 + k);
    acc0 = __builtin_amdgcn_wmma_f32_16x16x32_bf16(false, a, false, w0, (short)0, acc0, false, false);
    acc1 = __builtin_amdgcn_wmma_f32_16x16x32_bf16(false, a, false, w1, (short)0, acc1, false, false);
  }
  epi_store(acc0, oc0 + col,      half * 8, bias, bng, bnb, bnm, bnv, resid, dst, dst2, act);
  epi_store(acc1, oc0 + 16 + col, half * 8, bias, bng, bnb, bnm, bnv, resid, dst, dst2, act);
}

// A-panel builders -----------------------------------------------------------
__device__ __forceinline__ void build_ap(unsigned short* AP, const unsigned short* SRC, bool act) {
  for (int i = threadIdx.x; i < 16 * 256; i += 256) {
    unsigned short u = SRC[i];
    if (act) u = f2bf(nonlu(bf2f(u)));
    AP[i] = u;
  }
}
// interleave_jd: K-vector order c*3+t : (j[2k], j[2k+1], d[k])
__device__ __forceinline__ void build_ib_jd(unsigned short* IB, const unsigned short* J,
                                            const unsigned short* D, int k, bool act) {
  for (int i = threadIdx.x; i < 16 * 256; i += 256) {
    int ev = i >> 8, c = i & 255;
    float a0 = bf2f(J[((2 * k + 0) * 16 + ev) * 256 + c]);
    float a1 = bf2f(J[((2 * k + 1) * 16 + ev) * 256 + c]);
    float a2 = bf2f(D[(k * 16 + ev) * 256 + c]);
    if (act) { a0 = nonlu(a0); a1 = nonlu(a1); a2 = nonlu(a2); }
    unsigned short* o = IB + ev * 768 + c * 3;
    o[0] = f2bf(a0); o[1] = f2bf(a1); o[2] = f2bf(a2);
  }
}
// interleave_dq: (d[2k], d[2k+1], q[k]); optional nonlu on q only
__device__ __forceinline__ void build_ib_dq(unsigned short* IB, const unsigned short* D,
                                            const unsigned short* Q, int k, bool actQ) {
  for (int i = threadIdx.x; i < 16 * 256; i += 256) {
    int ev = i >> 8, c = i & 255;
    float a0 = bf2f(D[((2 * k + 0) * 16 + ev) * 256 + c]);
    float a1 = bf2f(D[((2 * k + 1) * 16 + ev) * 256 + c]);
    float a2 = bf2f(Q[(k * 16 + ev) * 256 + c]);
    if (actQ) a2 = nonlu(a2);
    unsigned short* o = IB + ev * 768 + c * 3;
    o[0] = f2bf(a0); o[1] = f2bf(a1); o[2] = f2bf(a2);
  }
}

// ---------------------------------------------------------------------------
struct Params {
  const float* jin; long n;
  const float *cj_m, *cj_s, *dj_m, *dj_s, *qj_m, *qj_s;
  const float *ptG, *ptB, *ptM, *ptV;
  const float *etG, *etB, *etM, *etV;
  const float *mG, *mB, *mM, *mV;
  const float *dG, *dB, *dM, *dV;
  const float *qG, *qB, *qM, *qV;
  const float *jeW, *jeb, *de1W, *de1b, *qeW, *qeb;
  const float *de2b;
  const float *dr1b, *dr1g, *dr1bb, *dr1m, *dr1v;
  const float *cJb,  *cJg,  *cJbb,  *cJm,  *cJv;
  const float *dr2b;
  const float *qr1b, *qr1g, *qr1bb, *qr1m, *qr1v;
  const float *cDb,  *cDg,  *cDbb,  *cDm,  *cDv;
  const float *qr2b;
  const float *ec1b, *ec1g, *ec1bb, *ec1m, *ec1v;
  const float *ec2b;
  const float *selW, *selb, *selg, *selbb, *selm, *selv;
  const unsigned short *Wde2, *Wdr1, *WcJ, *Wdr2, *Wqr1, *WcD, *Wqr2, *Wec1, *Wec2;
  float* out;
};

__global__ void convert_f32_bf16(const float* __restrict__ src,
                                 unsigned short* __restrict__ dst, int n) {
  int i = blockIdx.x * 256 + threadIdx.x;
  if (i < n) dst[i] = f2bf(src[i]);
}

__global__ void __launch_bounds__(256, 1) fvt_fused_kernel(Params P) {
  extern __shared__ __align__(16) unsigned char smem_raw[];
  unsigned short* J  = (unsigned short*)(smem_raw);            // 192 tok x 256 (je / j)
  unsigned short* D  = (unsigned short*)(smem_raw + 98304);    // 96 tok x 256
  unsigned short* D0 = (unsigned short*)(smem_raw + 147456);   // residual d0
  unsigned short* Q  = (unsigned short*)(smem_raw + 196608);   // 48 tok x 256
  unsigned short* Q0 = (unsigned short*)(smem_raw + 221184);   // residual q0
  unsigned short* IB = (unsigned short*)(smem_raw + 245760);   // A-panel 16 x 768
  float* sJn = (float*)(smem_raw + 270336);                    // 192 x 4
  float* sDn = (float*)(smem_raw + 273408);                    // 96 x 2
  float* sQn = (float*)(smem_raw + 274176);                    // 48 x 2
  float* sSc = (float*)(smem_raw + 274560);                    // 48 scores
  float* sWt = (float*)(smem_raw + 274752);                    // 48 softmax weights

  // ---- phase 0: per-event ancillary physics features + input norms ----
  if (threadIdx.x < 16) {
    int  ev = threadIdx.x;
    long g  = (long)blockIdx.x * 16 + ev;
    if (g < P.n) {
      const float* jp = P.jin + g * 16;
      float pt[4], et[4], ph[4], ms[4], px[4], py[4], pz[4], en[4];
#pragma unroll
      for (int i = 0; i < 4; ++i) { pt[i]=jp[i]; et[i]=jp[4+i]; ph[i]=jp[8+i]; ms[i]=jp[12+i]; }
#pragma unroll
      for (int i = 0; i < 4; ++i) {
        px[i] = pt[i] * cosf(ph[i]); py[i] = pt[i] * sinf(ph[i]); pz[i] = pt[i] * sinhf(et[i]);
        en[i] = sqrtf(ms[i]*ms[i] + px[i]*px[i] + py[i]*py[i] + pz[i]*pz[i]);
      }
      const int PA[6] = {0,2,0,1,0,1}, PB[6] = {1,3,2,3,3,2};
      float dpx[6], dpy[6], dpz[6], de[6], dmv[6], ddr[6], dEta[6], dPhi[6];
#pragma unroll
      for (int p = 0; p < 6; ++p) {
        int a = PA[p], b = PB[p];
        dpx[p]=px[a]+px[b]; dpy[p]=py[a]+py[b]; dpz[p]=pz[a]+pz[b]; de[p]=en[a]+en[b];
        float m2 = de[p]*de[p]-dpx[p]*dpx[p]-dpy[p]*dpy[p]-dpz[p]*dpz[p];
        dmv[p] = sqrtf(fmaxf(m2, 1e-12f));
        float deta = et[a]-et[b], dphw = dphi_wrap(ph[a]-ph[b]);
        ddr[p] = sqrtf(deta*deta + dphw*dphw);
        float dpt = sqrtf(dpx[p]*dpx[p]+dpy[p]*dpy[p]) + 1e-12f;
        dEta[p] = asinhf(dpz[p]/dpt);
        dPhi[p] = atan2f(dpy[p], dpx[p]);
      }
      float qmv[3], qdr[3];
#pragma unroll
      for (int s = 0; s < 3; ++s) {
        int a = 2*s, b = 2*s+1;
        float qe=de[a]+de[b], qx=dpx[a]+dpx[b], qy=dpy[a]+dpy[b], qz=dpz[a]+dpz[b];
        qmv[s] = sqrtf(fmaxf(qe*qe-qx*qx-qy*qy-qz*qz, 1e-12f));
        float e1 = dEta[a]-dEta[b], p1 = dphi_wrap(dPhi[a]-dPhi[b]);
        qdr[s] = sqrtf(e1*e1 + p1*p1);
      }
      // fold 1/2-channel gbn into scale/shift
      float ptS=P.ptG[0]/sqrtf(P.ptV[0]+EPSV), ptSh=P.ptB[0]-P.ptM[0]*ptS;
      float etS=P.etG[0]/sqrtf(P.etV[0]+EPSV), etSh=P.etB[0]-P.etM[0]*etS;
      float mS =P.mG[0] /sqrtf(P.mV[0] +EPSV), mSh =P.mB[0] -P.mM[0] *mS;
      float dS0=P.dG[0]/sqrtf(P.dV[0]+EPSV), dSh0=P.dB[0]-P.dM[0]*dS0;
      float dS1=P.dG[1]/sqrtf(P.dV[1]+EPSV), dSh1=P.dB[1]-P.dM[1]*dS1;
      float qS0=P.qG[0]/sqrtf(P.qV[0]+EPSV), qSh0=P.qB[0]-P.qM[0]*qS0;
      float qS1=P.qG[1]/sqrtf(P.qV[1]+EPSV), qSh1=P.qB[1]-P.qM[1]*qS1;
      const int J12[12] = {0,1,2,3,0,2,1,3,0,3,1,2};
#pragma unroll
      for (int p = 0; p < 12; ++p) {
        int jj = J12[p];
        float* o = sJn + (p * 16 + ev) * 4;
        o[0] = (pt[jj]-P.cj_m[0])/P.cj_s[0]*ptS + ptSh;
        o[1] = (et[jj]-P.cj_m[1])/P.cj_s[1]*etS + etSh;
        o[2] = (ph[jj]-P.cj_m[2])/P.cj_s[2];
        o[3] = (ms[jj]-P.cj_m[3])/P.cj_s[3]*mS + mSh;
      }
#pragma unroll
      for (int p = 0; p < 6; ++p) {
        sDn[(p*16+ev)*2+0] = (dmv[p]-P.dj_m[0])/P.dj_s[0]*dS0 + dSh0;
        sDn[(p*16+ev)*2+1] = (ddr[p]-P.dj_m[1])/P.dj_s[1]*dS1 + dSh1;
      }
#pragma unroll
      for (int s = 0; s < 3; ++s) {
        sQn[(s*16+ev)*2+0] = (qmv[s]-P.qj_m[0])/P.qj_s[0]*qS0 + qSh0;
        sQn[(s*16+ev)*2+1] = (qdr[s]-P.qj_m[1])/P.qj_s[1]*qS1 + qSh1;
      }
    } else {  // pad events: zero so GEMMs stay finite
      for (int p = 0; p < 12; ++p) { float* o = sJn + (p*16+ev)*4; o[0]=o[1]=o[2]=o[3]=0.f; }
      for (int p = 0; p < 6;  ++p) { sDn[(p*16+ev)*2]=0.f; sDn[(p*16+ev)*2+1]=0.f; }
      for (int s = 0; s < 3;  ++s) { sQn[(s*16+ev)*2]=0.f; sQn[(s*16+ev)*2+1]=0.f; }
    }
  }
  __syncthreads();

  // ---- tiny-K embeddings (VALU): je(K=4), de1(K=2), qe(K=2) ----
  {
    int oc = threadIdx.x;
    float w0=P.jeW[oc*4+0], w1=P.jeW[oc*4+1], w2=P.jeW[oc*4+2], w3=P.jeW[oc*4+3], jb=P.jeb[oc];
    for (int t = 0; t < 192; ++t) {
      const float* x = sJn + t * 4;
      J[t*256+oc] = f2bf(jb + w0*x[0] + w1*x[1] + w2*x[2] + w3*x[3]);
    }
    float e0=P.de1W[oc*2+0], e1=P.de1W[oc*2+1], eb=P.de1b[oc];
    for (int t = 0; t < 96; ++t) {
      unsigned short h = f2bf(eb + e0*sDn[t*2] + e1*sDn[t*2+1]);
      D[t*256+oc] = h; D0[t*256+oc] = h;
    }
    float g0=P.qeW[oc*2+0], g1=P.qeW[oc*2+1], gb=P.qeb[oc];
    for (int t = 0; t < 48; ++t) {
      unsigned short h = f2bf(gb + g0*sQn[t*2] + g1*sQn[t*2+1]);
      Q[t*256+oc] = h; Q0[t*256+oc] = h;
    }
  }
  __syncthreads();

  // ---- dijet block ----
#pragma unroll 1
  for (int k = 0; k < 6; ++k) {                 // r1: conv3s3 on nonlu(je),nonlu(d)
    build_ib_jd(IB, J, D, k, true);
    __syncthreads();
    gemm16(IB, 768, P.Wdr1, P.dr1b, P.dr1g, P.dr1bb, P.dr1m, P.dr1v,
           D0 + k*16*256, D + k*16*256, nullptr, true);
    __syncthreads();
  }
#pragma unroll 1
  for (int p = 0; p < 12; ++p) {                // cJ: conv1 on nonlu(je), resid je
    build_ap(IB, J + p*16*256, true);
    __syncthreads();
    gemm16(IB, 256, P.WcJ, P.cJb, P.cJg, P.cJbb, P.cJm, P.cJv,
           J + p*16*256, J + p*16*256, nullptr, true);
    __syncthreads();
  }
#pragma unroll 1
  for (int k = 0; k < 6; ++k) {                 // r2: conv3s3; result also becomes new d0
    build_ib_jd(IB, J, D, k, false);
    __syncthreads();
    gemm16(IB, 768, P.Wdr2, P.dr2b, nullptr, nullptr, nullptr, nullptr,
           D0 + k*16*256, D + k*16*256, D0 + k*16*256, true);
    __syncthreads();
  }
  // ---- de2 + residual ----
#pragma unroll 1
  for (int p = 0; p < 6; ++p) {
    build_ap(IB, D + p*16*256, false);
    __syncthreads();
    gemm16(IB, 256, P.Wde2, P.de2b, nullptr, nullptr, nullptr, nullptr,
           D0 + p*16*256, D + p*16*256, nullptr, true);
    __syncthreads();
  }
  // ---- quadjet block ----
#pragma unroll 1
  for (int k = 0; k < 3; ++k) {                 // r1: conv3s3 on (d, nonlu(q_embed))
    build_ib_dq(IB, D, Q, k, true);
    __syncthreads();
    gemm16(IB, 768, P.Wqr1, P.qr1b, P.qr1g, P.qr1bb, P.qr1m, P.qr1v,
           Q0 + k*16*256, Q + k*16*256, nullptr, true);
    __syncthreads();
  }
#pragma unroll 1
  for (int p = 0; p < 6; ++p) {                 // cD: conv1 on d
    build_ap(IB, D + p*16*256, false);
    __syncthreads();
    gemm16(IB, 256, P.WcD, P.cDb, P.cDg, P.cDbb, P.cDm, P.cDv,
           D0 + p*16*256, D + p*16*256, nullptr, true);
    __syncthreads();
  }
#pragma unroll 1
  for (int k = 0; k < 3; ++k) {                 // r2: result also becomes new q0
    build_ib_dq(IB, D, Q, k, false);
    __syncthreads();
    gemm16(IB, 768, P.Wqr2, P.qr2b, nullptr, nullptr, nullptr, nullptr,
           Q0 + k*16*256, Q + k*16*256, Q0 + k*16*256, true);
    __syncthreads();
  }
  // ---- ec1 / ec2 ----
#pragma unroll 1
  for (int k = 0; k < 3; ++k) {
    build_ap(IB, Q + k*16*256, false);
    __syncthreads();
    gemm16(IB, 256, P.Wec1, P.ec1b, P.ec1g, P.ec1bb, P.ec1m, P.ec1v,
           Q0 + k*16*256, Q + k*16*256, nullptr, true);
    __syncthreads();
  }
#pragma unroll 1
  for (int k = 0; k < 3; ++k) {
    build_ap(IB, Q + k*16*256, false);
    __syncthreads();
    gemm16(IB, 256, P.Wec2, P.ec2b, nullptr, nullptr, nullptr, nullptr,
           Q0 + k*16*256, Q + k*16*256, nullptr, true);
    __syncthreads();
  }

  // ---- selection scores + softmax attention pooling ----
  if (threadIdx.x < 48) {
    int ev = threadIdx.x & 15, s = threadIdx.x >> 4;
    float acc = P.selb[0];
    const unsigned short* qrow = Q + (s * 16 + ev) * 256;
    for (int c = 0; c < 256; ++c) acc += bf2f(qrow[c]) * P.selW[c];
    float sc = P.selg[0] / sqrtf(P.selv[0] + EPSV);
    sSc[s * 16 + ev] = (acc - P.selm[0]) * sc + P.selbb[0];
  }
  __syncthreads();
  if (threadIdx.x < 16) {
    int ev = threadIdx.x;
    float s0 = sSc[ev], s1 = sSc[16 + ev], s2 = sSc[32 + ev];
    float mx = fmaxf(s0, fmaxf(s1, s2));
    float e0 = __expf(s0 - mx), e1 = __expf(s1 - mx), e2 = __expf(s2 - mx);
    float inv = 1.f / (e0 + e1 + e2);
    sWt[ev] = e0 * inv; sWt[16 + ev] = e1 * inv; sWt[32 + ev] = e2 * inv;
  }
  __syncthreads();
  for (int i = threadIdx.x; i < 16 * 256; i += 256) {
    int ev = i >> 8, c = i & 255;
    long g = (long)blockIdx.x * 16 + ev;
    if (g < P.n) {
      float acc = bf2f(Q[(0*16+ev)*256+c]) * sWt[ev]
                + bf2f(Q[(1*16+ev)*256+c]) * sWt[16+ev]
                + bf2f(Q[(2*16+ev)*256+c]) * sWt[32+ev];
      P.out[g * 256 + c] = acc;
    }
  }
}

// ---------------------------------------------------------------------------
// d_in flattening (setup_inputs dict order, depth-first):
//  0 j | 1 cj_m 2 cj_s 3 dj_m 4 dj_s 5 qj_m 6 qj_s
//  7-10 pt_bn(g,b,m,v) 11-14 eta_bn 15-18 m_bn 19-22 d_bn 23-26 q_bn
//  27 je_W 28 je_b 29 de1_W 30 de1_b 31 qe_W 32 qe_b 33 de2_W 34 de2_b
//  dblock: 35 r1_W 36 r1_b 37-40 r1_bn 41 cJ_W 42 cJ_b 43-46 cJ_bn 47 r2_W 48 r2_b
//  qblock: 49 r1_W 50 r1_b 51-54 r1_bn 55 cD_W 56 cD_b 57-60 cD_bn 61 r2_W 62 r2_b
//  63 ec1_W 64 ec1_b 65-68 ec1_bn 69 ec2_W 70 ec2_b 71 sel_W 72 sel_b 73-76 sel_bn
// ---------------------------------------------------------------------------
extern "C" void kernel_launch(void* const* d_in, const int* in_sizes, int n_in,
                              void* d_out, int out_size, void* d_ws, size_t ws_size,
                              hipStream_t stream) {
  auto F = [&](int i) { return (const float*)d_in[i]; };
  long n = in_sizes[0] / 16;

  unsigned short* ws   = (unsigned short*)d_ws;
  unsigned short* Wde2 = ws + 0;         // 65536
  unsigned short* Wdr1 = ws + 65536;     // 196608
  unsigned short* WcJ  = ws + 262144;    // 65536
  unsigned short* Wdr2 = ws + 327680;    // 196608
  unsigned short* Wqr1 = ws + 524288;    // 196608
  unsigned short* WcD  = ws + 720896;    // 65536
  unsigned short* Wqr2 = ws + 786432;    // 196608
  unsigned short* Wec1 = ws + 983040;    // 65536
  unsigned short* Wec2 = ws + 1048576;   // 65536  (total 2228224 bytes)

  auto conv = [&](const float* s, unsigned short* d, int cnt) {
    convert_f32_bf16<<<(cnt + 255) / 256, 256, 0, stream>>>(s, d, cnt);
  };
  conv(F(33), Wde2, 65536);
  conv(F(35), Wdr1, 196608);
  conv(F(41), WcJ,  65536);
  conv(F(47), Wdr2, 196608);
  conv(F(49), Wqr1, 196608);
  conv(F(55), WcD,  65536);
  conv(F(61), Wqr2, 196608);
  conv(F(63), Wec1, 65536);
  conv(F(69), Wec2, 65536);

  Params P;
  P.jin = F(0); P.n = n;
  P.cj_m = F(1); P.cj_s = F(2); P.dj_m = F(3); P.dj_s = F(4); P.qj_m = F(5); P.qj_s = F(6);
  P.ptG = F(7);  P.ptB = F(8);  P.ptM = F(9);  P.ptV = F(10);
  P.etG = F(11); P.etB = F(12); P.etM = F(13); P.etV = F(14);
  P.mG  = F(15); P.mB  = F(16); P.mM  = F(17); P.mV  = F(18);
  P.dG  = F(19); P.dB  = F(20); P.dM  = F(21); P.dV  = F(22);
  P.qG  = F(23); P.qB  = F(24); P.qM  = F(25); P.qV  = F(26);
  P.jeW = F(27); P.jeb = F(28); P.de1W = F(29); P.de1b = F(30); P.qeW = F(31); P.qeb = F(32);
  P.de2b = F(34);
  P.dr1b = F(36); P.dr1g = F(37); P.dr1bb = F(38); P.dr1m = F(39); P.dr1v = F(40);
  P.cJb  = F(42); P.cJg  = F(43); P.cJbb  = F(44); P.cJm  = F(45); P.cJv  = F(46);
  P.dr2b = F(48);
  P.qr1b = F(50); P.qr1g = F(51); P.qr1bb = F(52); P.qr1m = F(53); P.qr1v = F(54);
  P.cDb  = F(56); P.cDg  = F(57); P.cDbb  = F(58); P.cDm  = F(59); P.cDv  = F(60);
  P.qr2b = F(62);
  P.ec1b = F(64); P.ec1g = F(65); P.ec1bb = F(66); P.ec1m = F(67); P.ec1v = F(68);
  P.ec2b = F(70);
  P.selW = F(71); P.selb = F(72); P.selg = F(73); P.selbb = F(74); P.selm = F(75); P.selv = F(76);
  P.Wde2 = Wde2; P.Wdr1 = Wdr1; P.WcJ = WcJ; P.Wdr2 = Wdr2;
  P.Wqr1 = Wqr1; P.WcD = WcD; P.Wqr2 = Wqr2; P.Wec1 = Wec1; P.Wec2 = Wec2;
  P.out = (float*)d_out;

  hipFuncSetAttribute((const void*)fvt_fused_kernel,
                      hipFuncAttributeMaxDynamicSharedMemorySize, SMEM_BYTES);
  int blocks = (int)((n + 15) / 16);
  fvt_fused_kernel<<<blocks, 256, SMEM_BYTES, stream>>>(P);
}
